// MPNN_63024350101880
// MI455X (gfx1250) — matrix operand order repaired and checked by reading, not compiled, not run on hardware
//
#include <hip/hip_runtime.h>
#include <hip/hip_bf16.h>

// ---------------- constants ----------------
constexpr int N_  = 100000;
constexpr int E_  = 1600000;
constexpr int L_  = 4;
constexpr int G_  = 64;
constexpr float EPS_ = 1e-5f;

static_assert(E_ % 32 == 0, "edge tiles");
static_assert(N_ % 16 == 0, "node tiles");

// ---------------- vector types for WMMA ----------------
typedef __bf16  v16bf __attribute__((ext_vector_type(16)));
typedef float   v8f   __attribute__((ext_vector_type(8)));
typedef unsigned int v4u __attribute__((ext_vector_type(4)));
typedef float   v4f   __attribute__((ext_vector_type(4)));

#define WMMA_BF16(A, B, C) \
  __builtin_amdgcn_wmma_f32_16x16x32_bf16(false, (A), false, (B), (short)0, (C), false, false)

__device__ __forceinline__ void lds_fence() {
  // drain LDS ops so data written by other lanes of THIS wave is visible
  asm volatile("s_wait_dscnt 0x0" ::: "memory");
}

// Load a 16x32 bf16 A-fragment from a row-major buffer (stride in elements).
// Layout per CDNA5 ISA 7.12.2: lanes 0-15 hold row M=lane, K in {0..7,16..23};
// lanes 16-31 hold row M=lane-16, K in {8..15,24..31}. Two b128 loads per lane.
__device__ __forceinline__ v16bf load_afrag(const __bf16* base, int stride, int k0, int lane) {
  int row = lane & 15;
  int kh  = (lane >> 4) << 3;            // 0 or 8
  union { v4u u[2]; v16bf v; } t;
  t.u[0] = *(const v4u*)(base + (size_t)row * stride + k0 + kh);
  t.u[1] = *(const v4u*)(base + (size_t)row * stride + k0 + kh + 16);
  return t.v;
}

// B fragment: pre-swizzled so each lane's 16 bf16 are contiguous (32B).
__device__ __forceinline__ v16bf load_bfrag(const __bf16* w, int frag, int lane) {
  return *(const v16bf*)(w + (size_t)(frag * 32 + lane) * 16);
}

// ---------------- weight prep ----------------
// Re-pack a [L][Ksrc][128] f32 weight into WMMA-B-fragment order:
// frag(ks,nt) of 32 lanes x 16 bf16 contiguous. Lane L: col = nt*16+(L&15),
// kb = (L>>4)*8, element e -> K = ks*32 + kb + (e<8 ? e : e+8). Optional BN scale on K row.
__global__ void prep_frag_kernel(const float* __restrict__ W,
                                 const float* __restrict__ bn_g,
                                 const float* __restrict__ bn_v,
                                 __bf16* __restrict__ out,
                                 int Ksrc, int Kpad, int total) {
  int t = blockIdx.x * blockDim.x + threadIdx.x;
  if (t >= total) return;
  int perLayer = Kpad * 128;
  int l = t / perLayer;
  int r = t - l * perLayer;
  int frag = r >> 9;            // / 512
  int lane = (r >> 4) & 31;
  int e    = r & 15;
  int ks = frag >> 3;
  int nt = frag & 7;
  int col = nt * 16 + (lane & 15);
  int kb  = (lane >> 4) << 3;   // 0 or 8
  int K   = ks * 32 + kb + (e < 8 ? e : e + 8);
  float val = 0.f;
  if (K < Ksrc) {
    val = W[(size_t)l * Ksrc * 128 + (size_t)K * 128 + col];
    if (bn_g) {
      float s = bn_g[l * 128 + K] * rsqrtf(bn_v[l * 128 + K] + EPS_);
      val *= s;
    }
  }
  out[t] = (__bf16)val;
}

// Fold BN shift into second-linear bias: b2'[d] = b2[d] + sum_c (bnb - bnm*s)[c] * W2[c][d]
__global__ void prep_bias_kernel(const float* __restrict__ W2, const float* __restrict__ b2,
                                 const float* __restrict__ g,  const float* __restrict__ bb,
                                 const float* __restrict__ m,  const float* __restrict__ v,
                                 float* __restrict__ out, int total) {
  int t = blockIdx.x * blockDim.x + threadIdx.x;
  if (t >= total) return;
  int l = t >> 7;
  int d = t & 127;
  const float* Wl = W2 + (size_t)l * 128 * 128;
  float acc = b2[t];
  for (int c = 0; c < 128; ++c) {
    float s  = g[l * 128 + c] * rsqrtf(v[l * 128 + c] + EPS_);
    float tt = bb[l * 128 + c] - m[l * 128 + c] * s;
    acc += tt * Wl[c * 128 + d];
  }
  out[t] = acc;
}

// ---------------- input embedding ----------------
__global__ __launch_bounds__(128)
void lin_in_kernel(const float* __restrict__ x, const float* __restrict__ pos,
                   const float* __restrict__ w, const float* __restrict__ b,
                   float* __restrict__ h32, __bf16* __restrict__ hbf) {
  int n = blockIdx.x;
  int d = threadIdx.x;
  __shared__ float in_s[14];
  if (d < 11)       in_s[d] = x[(size_t)n * 11 + d];
  else if (d < 14)  in_s[d] = pos[(size_t)n * 3 + (d - 11)];
  __syncthreads();
  float acc = b[d];
#pragma unroll
  for (int k = 0; k < 14; ++k) acc += in_s[k] * w[k * 128 + d];
  size_t o = (size_t)n * 128 + d;
  h32[o] = acc;
  hbf[o] = (__bf16)acc;
}

// ---------------- message + scatter-add ----------------
// One wave per 32-edge tile (M=32: two A tiles share each B fragment, halving
// weight traffic vs M=16). N dimension is processed in two halves of 4 output
// tiles so accumulator pressure stays at 64 VGPRs (no VGPR-MSB mode); B traffic
// is unchanged by the split since each (ks,nt) fragment is still loaded once.
__global__ __launch_bounds__(64)
void msg_kernel(const __bf16* __restrict__ hbf,
                const int* __restrict__ srcI, const int* __restrict__ dstI,
                const float* __restrict__ ea,
                const __bf16* __restrict__ w1, const float* __restrict__ b1,
                const __bf16* __restrict__ w2, const float* __restrict__ b2,
                float* __restrict__ aggr, int nTiles) {
  constexpr int A1S = 296;   // padded row stride (elements) to dodge bank conflicts
  constexpr int M1S = 136;
  __shared__ __bf16 A1s[2][32 * A1S];
  __shared__ __bf16 M1s[2][32 * M1S];
  __shared__ int dIs[2][32];
  __shared__ int sIs[2][32];

  int wave = threadIdx.x >> 5;
  int lane = threadIdx.x & 31;
  int tile = blockIdx.x * 2 + wave;
  if (tile >= nTiles) return;

  __bf16* A1 = A1s[wave];
  __bf16* M1 = M1s[wave];
  int* dI = dIs[wave];
  int* sI = sIs[wave];
  int e0 = tile * 32;

  dI[lane] = dstI[e0 + lane];
  sI[lane] = srcI[e0 + lane];
  lds_fence();

  // gather h[dst] | h[src] into LDS rows; one row per iter, 16B chunk per lane
#pragma unroll 4
  for (int row = 0; row < 32; ++row) {
    int node = (lane < 16) ? dI[row] : sI[row];
    int off  = (lane & 15) * 8;
    int doff = row * A1S + ((lane < 16) ? 0 : 128) + off;
    *(v4u*)(A1 + doff) = *(const v4u*)(hbf + (size_t)node * 128 + off);
  }
  {
    int row = lane;
#pragma unroll
    for (int k = 0; k < 4; ++k)
      A1[row * A1S + 256 + k] = (__bf16)ea[(size_t)(e0 + row) * 4 + k];
#pragma unroll
    for (int k = 260; k < 288; ++k)
      A1[row * A1S + k] = (__bf16)0.0f;
  }
  lds_fence();

  const v8f vzero = {0.f, 0.f, 0.f, 0.f, 0.f, 0.f, 0.f, 0.f};
  int col   = lane & 15;
  int rbase = (lane >> 4) << 3;

  // ---- GEMM1: [32x288] x [288x128], two N-halves of 4 tiles each ----
  for (int nh = 0; nh < 2; ++nh) {
    const int fb = nh * 4;               // first n-tile of this half
    v8f accL[4], accH[4];
#pragma unroll
    for (int t = 0; t < 4; ++t) { accL[t] = vzero; accH[t] = vzero; }

    v16bf bcur = load_bfrag(w1, fb, lane);
    for (int ks = 0; ks < 9; ++ks) {
      v16bf a0 = load_afrag(A1, A1S, ks * 32, lane);
      v16bf a1 = load_afrag(A1 + 16 * A1S, A1S, ks * 32, lane);
#pragma unroll
      for (int t = 0; t < 4; ++t) {
        int nf = (t < 3) ? (ks * 8 + fb + t + 1) : ((ks + 1) * 8 + fb);
        v16bf bnext = load_bfrag(w1, nf, lane);   // pad frags make this safe
        accL[t] = WMMA_BF16(a0, bcur, accL[t]);
        accH[t] = WMMA_BF16(a1, bcur, accH[t]);
        bcur = bnext;
      }
    }
    // bias + ReLU -> bf16 staging for GEMM2
#pragma unroll
    for (int t = 0; t < 4; ++t) {
      int nt = fb + t;
      float bias = b1[nt * 16 + col];
#pragma unroll
      for (int r = 0; r < 8; ++r) {
        float v0 = accL[t][r] + bias; v0 = v0 > 0.f ? v0 : 0.f;
        float v1 = accH[t][r] + bias; v1 = v1 > 0.f ? v1 : 0.f;
        M1[(rbase + r) * M1S + nt * 16 + col]      = (__bf16)v0;
        M1[(rbase + r + 16) * M1S + nt * 16 + col] = (__bf16)v1;
      }
    }
  }
  lds_fence();

  // ---- GEMM2 (BN folded): [32x128] x [128x128], two N-halves ----
  for (int nh = 0; nh < 2; ++nh) {
    const int fb = nh * 4;
    v8f accL[4], accH[4];
#pragma unroll
    for (int t = 0; t < 4; ++t) { accL[t] = vzero; accH[t] = vzero; }

    v16bf bcur = load_bfrag(w2, fb, lane);
    for (int ks = 0; ks < 4; ++ks) {
      v16bf a0 = load_afrag(M1, M1S, ks * 32, lane);
      v16bf a1 = load_afrag(M1 + 16 * M1S, M1S, ks * 32, lane);
#pragma unroll
      for (int t = 0; t < 4; ++t) {
        int nf = (t < 3) ? (ks * 8 + fb + t + 1) : ((ks + 1) * 8 + fb);
        v16bf bnext = load_bfrag(w2, nf, lane);
        accL[t] = WMMA_BF16(a0, bcur, accL[t]);
        accH[t] = WMMA_BF16(a1, bcur, accH[t]);
        bcur = bnext;
      }
    }
    // bias + ReLU + scatter-add into aggregation buffer
#pragma unroll
    for (int t = 0; t < 4; ++t) {
      int nt = fb + t;
      float bias = b2[nt * 16 + col];
#pragma unroll
      for (int r = 0; r < 8; ++r) {
        float v0 = accL[t][r] + bias; v0 = v0 > 0.f ? v0 : 0.f;
        float v1 = accH[t][r] + bias; v1 = v1 > 0.f ? v1 : 0.f;
        atomicAdd(&aggr[(size_t)dI[rbase + r] * 128 + nt * 16 + col], v0);
        atomicAdd(&aggr[(size_t)dI[rbase + r + 16] * 128 + nt * 16 + col], v1);
      }
    }
  }
}

// ---------------- node update (residual MLP) ----------------
__global__ __launch_bounds__(128)
void upd_kernel(float* __restrict__ h32, __bf16* __restrict__ hbf,
                const float* __restrict__ aggr,
                const __bf16* __restrict__ w1, const float* __restrict__ b1,
                const __bf16* __restrict__ w2, const float* __restrict__ b2,
                int nTiles) {
  constexpr int S = 136;
  __shared__ __bf16 AGs[4][16 * S];
  __shared__ __bf16 M1s[4][16 * S];

  int wave = threadIdx.x >> 5;
  int lane = threadIdx.x & 31;
  int tile = blockIdx.x * 4 + wave;
  if (tile >= nTiles) return;

  __bf16* AG = AGs[wave];
  __bf16* M1 = M1s[wave];
  int nb = tile * 16;

  // stage aggr (f32) -> bf16 LDS
#pragma unroll
  for (int it = 0; it < 16; ++it) {
    int idx = lane + it * 32;
    int row = idx >> 5;
    int j   = idx & 31;              // chunk of 4 floats
    v4f f = *(const v4f*)(aggr + (size_t)(nb + row) * 128 + j * 4);
    __bf16* o = AG + row * S + j * 4;
    o[0] = (__bf16)f[0]; o[1] = (__bf16)f[1]; o[2] = (__bf16)f[2]; o[3] = (__bf16)f[3];
  }
  lds_fence();

  const v8f vzero = {0.f, 0.f, 0.f, 0.f, 0.f, 0.f, 0.f, 0.f};
  v8f acc[8];
#pragma unroll
  for (int t = 0; t < 8; ++t) acc[t] = vzero;

  // GEMM1: K = 256 (first 128 = h from global bf16, last 128 = aggr from LDS)
  v16bf bcur = load_bfrag(w1, 0, lane);
  for (int ks = 0; ks < 8; ++ks) {
    int k0 = ks * 32;
    v16bf a;
    if (k0 < 128) {
      int row = lane & 15;
      int kh  = (lane >> 4) << 3;
      union { v4u u[2]; v16bf v; } tu;
      tu.u[0] = *(const v4u*)(hbf + (size_t)(nb + row) * 128 + k0 + kh);
      tu.u[1] = *(const v4u*)(hbf + (size_t)(nb + row) * 128 + k0 + kh + 16);
      a = tu.v;
    } else {
      a = load_afrag(AG, S, k0 - 128, lane);
    }
#pragma unroll
    for (int t = 0; t < 8; ++t) {
      v16bf bnext = load_bfrag(w1, ks * 8 + t + 1, lane);
      acc[t] = WMMA_BF16(a, bcur, acc[t]);
      bcur = bnext;
    }
  }

  int col   = lane & 15;
  int rbase = (lane >> 4) << 3;
#pragma unroll
  for (int t = 0; t < 8; ++t) {
    float bias = b1[t * 16 + col];
#pragma unroll
    for (int r = 0; r < 8; ++r) {
      float vv = acc[t][r] + bias;
      vv = vv > 0.f ? vv : 0.f;
      M1[(rbase + r) * S + t * 16 + col] = (__bf16)vv;
    }
  }
  lds_fence();

#pragma unroll
  for (int t = 0; t < 8; ++t) acc[t] = vzero;

  // GEMM2 (BN folded): K = 128
  bcur = load_bfrag(w2, 0, lane);
  for (int ks = 0; ks < 4; ++ks) {
    v16bf a = load_afrag(M1, S, ks * 32, lane);
#pragma unroll
    for (int t = 0; t < 8; ++t) {
      v16bf bnext = load_bfrag(w2, ks * 8 + t + 1, lane);
      acc[t] = WMMA_BF16(a, bcur, acc[t]);
      bcur = bnext;
    }
  }

  // ReLU + residual, write h in f32 and bf16
#pragma unroll
  for (int t = 0; t < 8; ++t) {
    float bias = b2[t * 16 + col];
#pragma unroll
    for (int r = 0; r < 8; ++r) {
      float vv = acc[t][r] + bias;
      vv = vv > 0.f ? vv : 0.f;
      int m = rbase + r;
      size_t o = (size_t)(nb + m) * 128 + t * 16 + col;
      float nh = h32[o] + vv;
      h32[o] = nh;
      hbf[o] = (__bf16)nh;
    }
  }
}

// ---------------- pooling + prediction ----------------
__global__ void pool_accum_kernel(const float* __restrict__ h, const int* __restrict__ batch,
                                  float* __restrict__ pool, int total) {
  int t = blockIdx.x * blockDim.x + threadIdx.x;
  if (t >= total) return;
  int n = t >> 7;
  int d = t & 127;
  atomicAdd(&pool[(size_t)batch[n] * 128 + d], h[t]);
}

__global__ void count_accum_kernel(const int* __restrict__ batch, float* __restrict__ cnt, int n_) {
  int n = blockIdx.x * blockDim.x + threadIdx.x;
  if (n >= n_) return;
  atomicAdd(&cnt[batch[n]], 1.0f);
}

__global__ void pred_kernel(const float* __restrict__ pool, const float* __restrict__ cnt,
                            const float* __restrict__ w, const float* __restrict__ b,
                            float* __restrict__ out) {
  int g = threadIdx.x;
  if (g >= G_) return;
  float c = cnt[g];
  c = c > 1.f ? c : 1.f;
  float acc = b[0];
  for (int d = 0; d < 128; ++d) acc += (pool[g * 128 + d] / c) * w[d];
  out[g] = acc;
}

// ---------------- launch ----------------
extern "C" void kernel_launch(void* const* d_in, const int* in_sizes, int n_in,
                              void* d_out, int out_size, void* d_ws, size_t ws_size,
                              hipStream_t stream) {
  (void)in_sizes; (void)n_in; (void)out_size; (void)ws_size;
  const float* x       = (const float*)d_in[0];
  const float* pos     = (const float*)d_in[1];
  const int*   ei      = (const int*)d_in[2];
  const float* ea      = (const float*)d_in[3];
  const int*   batch   = (const int*)d_in[4];
  const float* lin_w   = (const float*)d_in[5];
  const float* lin_b   = (const float*)d_in[6];
  const float* msg_w1  = (const float*)d_in[7];
  const float* msg_b1  = (const float*)d_in[8];
  const float* msg_g   = (const float*)d_in[9];
  const float* msg_bb  = (const float*)d_in[10];
  const float* msg_m   = (const float*)d_in[11];
  const float* msg_v   = (const float*)d_in[12];
  const float* msg_w2  = (const float*)d_in[13];
  const float* msg_b2  = (const float*)d_in[14];
  const float* upd_w1  = (const float*)d_in[15];
  const float* upd_b1  = (const float*)d_in[16];
  const float* upd_g   = (const float*)d_in[17];
  const float* upd_bb  = (const float*)d_in[18];
  const float* upd_m   = (const float*)d_in[19];
  const float* upd_v   = (const float*)d_in[20];
  const float* upd_w2  = (const float*)d_in[21];
  const float* upd_b2  = (const float*)d_in[22];
  const float* pred_w  = (const float*)d_in[23];
  const float* pred_b  = (const float*)d_in[24];
  float* out = (float*)d_out;

  const int* srcI = ei;        // edge_index[0] (source j)
  const int* dstI = ei + E_;   // edge_index[1] (target i)

  // workspace carve-out (256B aligned); weight buffers padded by 8 dummy
  // fragments (4096 elems) so the B double-buffer prefetch is unconditional.
  constexpr size_t FPAD = 8 * 512;
  char* ws = (char*)d_ws;
  size_t off = 0;
  auto take = [&](size_t bytes) -> char* {
    char* p = ws + off;
    off = (off + bytes + 255) & ~(size_t)255;
    return p;
  };
  float*  h32  = (float*) take((size_t)N_ * 128 * 4);
  __bf16* hbf  = (__bf16*)take((size_t)N_ * 128 * 2);
  float*  aggr = (float*) take((size_t)N_ * 128 * 4);
  __bf16* mw1f = (__bf16*)take(((size_t)L_ * 288 * 128 + FPAD) * 2);
  __bf16* mw2f = (__bf16*)take(((size_t)L_ * 128 * 128 + FPAD) * 2);
  float*  mb2f = (float*) take((size_t)L_ * 128 * 4);
  __bf16* uw1f = (__bf16*)take(((size_t)L_ * 256 * 128 + FPAD) * 2);
  __bf16* uw2f = (__bf16*)take(((size_t)L_ * 128 * 128 + FPAD) * 2);
  float*  ub2f = (float*) take((size_t)L_ * 128 * 4);
  float*  pool = (float*) take((size_t)G_ * 128 * 4);
  float*  cnt  = (float*) take((size_t)G_ * 4);

  // weight prep: bf16 fragment packing + BN folding
  { int tot = L_ * 288 * 128;
    prep_frag_kernel<<<(tot + 255) / 256, 256, 0, stream>>>(msg_w1, nullptr, nullptr, mw1f, 260, 288, tot); }
  { int tot = L_ * 128 * 128;
    prep_frag_kernel<<<(tot + 255) / 256, 256, 0, stream>>>(msg_w2, msg_g, msg_v, mw2f, 128, 128, tot); }
  { int tot = L_ * 128;
    prep_bias_kernel<<<(tot + 255) / 256, 256, 0, stream>>>(msg_w2, msg_b2, msg_g, msg_bb, msg_m, msg_v, mb2f, tot); }
  { int tot = L_ * 256 * 128;
    prep_frag_kernel<<<(tot + 255) / 256, 256, 0, stream>>>(upd_w1, nullptr, nullptr, uw1f, 256, 256, tot); }
  { int tot = L_ * 128 * 128;
    prep_frag_kernel<<<(tot + 255) / 256, 256, 0, stream>>>(upd_w2, upd_g, upd_v, uw2f, 128, 128, tot); }
  { int tot = L_ * 128;
    prep_bias_kernel<<<(tot + 255) / 256, 256, 0, stream>>>(upd_w2, upd_b2, upd_g, upd_bb, upd_m, upd_v, ub2f, tot); }

  // input embedding
  lin_in_kernel<<<N_, 128, 0, stream>>>(x, pos, lin_w, lin_b, h32, hbf);

  const int eTiles = E_ / 32;   // 50000 (32-edge tiles)
  const int nTiles = N_ / 16;   // 6250
  for (int l = 0; l < L_; ++l) {
    hipMemsetAsync(aggr, 0, (size_t)N_ * 128 * 4, stream);
    msg_kernel<<<(eTiles + 1) / 2, 64, 0, stream>>>(
        hbf, srcI, dstI, ea,
        mw1f + (size_t)l * 288 * 128, msg_b1 + l * 128,
        mw2f + (size_t)l * 128 * 128, mb2f + l * 128,
        aggr, eTiles);
    upd_kernel<<<(nTiles + 3) / 4, 128, 0, stream>>>(
        h32, hbf, aggr,
        uw1f + (size_t)l * 256 * 128, upd_b1 + l * 128,
        uw2f + (size_t)l * 128 * 128, ub2f + l * 128,
        nTiles);
  }

  // global mean pool + linear head
  hipMemsetAsync(pool, 0, (size_t)G_ * 128 * 4, stream);
  hipMemsetAsync(cnt,  0, (size_t)G_ * 4, stream);
  { int tot = N_ * 128;
    pool_accum_kernel<<<(tot + 255) / 256, 256, 0, stream>>>(h32, batch, pool, tot); }
  count_accum_kernel<<<(N_ + 255) / 256, 256, 0, stream>>>(batch, cnt, N_);
  pred_kernel<<<1, 64, 0, stream>>>(pool, cnt, pred_w, pred_b, out);
}
